// TPLSTM_21895743275654
// MI455X (gfx1250) — matrix operand order repaired and checked by reading, not compiled, not run on hardware
//
#include <hip/hip_runtime.h>
#include <hip/hip_bf16.h>
#include <math.h>

// ---- problem dims (reference: S,B,F1,H = 256,256,257,512) ----
#define S_DIM  256
#define B_DIM  256
#define F_DIM  256
#define F1_DIM 257
#define H_DIM  512
#define G_DIM  (4 * H_DIM)          // 2048 gate columns
#define KG     (F_DIM + H_DIM)      // 768 packed contraction for gates

// WMMA fragment types (wave32, v_wmma_f32_16x16x32_bf16)
typedef __attribute__((ext_vector_type(16))) __bf16 v16bf;
typedef __attribute__((ext_vector_type(8)))  float  v8f;

static __device__ __forceinline__ unsigned short f2bf(float x) {
  unsigned u = __float_as_uint(x);
  u += 0x7FFFu + ((u >> 16) & 1u);          // RNE
  return (unsigned short)(u >> 16);
}

static __device__ __forceinline__ float fast_tanh(float x) {
#if __has_builtin(__builtin_amdgcn_tanhf)
  return __builtin_amdgcn_tanhf(x);         // native v_tanh_f32 (CDNA5 TRANS op)
#else
  float e = __expf(2.0f * x);
  return 1.0f - 2.0f / (e + 1.0f);
#endif
}

static __device__ __forceinline__ float fast_sigmoid(float x) {
#if __has_builtin(__builtin_amdgcn_rcpf)
  return __builtin_amdgcn_rcpf(1.0f + __expf(-x));
#else
  return 1.0f / (1.0f + __expf(-x));
#endif
}

// Load a 16-element bf16 WMMA fragment as two aligned 16B runs.
static __device__ __forceinline__ v16bf ld16(const unsigned short* base, int o1, int o2) {
  union { uint4 q[2]; v16bf v; } u;
  u.q[0] = *(const uint4*)(base + o1);
  u.q[1] = *(const uint4*)(base + o2);
  return u.v;
}

#define WMMA_BF16(A, B, C) \
  __builtin_amdgcn_wmma_f32_16x16x32_bf16(false, (A), false, (B), (short)0, (C), false, false)

// ---------------------------------------------------------------------------
// Generic f32 -> bf16 strided repack (prologue only, fully parallel).
// dst[r*dst_stride + dst_off + c] = bf16(src[r*src_stride + c]),  c < src_len
// ---------------------------------------------------------------------------
__global__ __launch_bounds__(256) void conv_bf16(
    const float* __restrict__ src, unsigned short* __restrict__ dst,
    int total, int src_stride, int src_len, int dst_stride, int dst_off)
{
  int i = (int)blockIdx.x * 256 + (int)threadIdx.x;
  if (i >= total) return;
  int r = i / src_len, c = i % src_len;
  dst[(size_t)r * dst_stride + dst_off + c] = f2bf(src[(size_t)r * src_stride + c]);
}

// ---------------------------------------------------------------------------
// Fused per-step GEMM. One 32x32 output block per wave (2x2 WMMA tiles):
//   waves 0..511  : pre[b,j] = [xi|h]_bf16 x wg_bf16^T   (M=256,N=2048,K=768)
//   waves 512..639: dec[b,j] = tanh(c_bf16 x wdec_bf16^T + b_dec) (M=256,N=512,K=512)
// All operands pre-converted to bf16 in WMMA-friendly contiguous rows, so the
// inner loop is pure global_load_b128 + v_wmma (no conversion VALU).
// ---------------------------------------------------------------------------
__global__ __launch_bounds__(256) void tplstm_gemm(
    const unsigned short* __restrict__ xibf,  // B x F   (this step's slice)
    const unsigned short* __restrict__ hbf,   // B x H
    const unsigned short* __restrict__ cbf,   // B x H
    const unsigned short* __restrict__ wg,    // G x KG  (w_ih || w_hh rows)
    const unsigned short* __restrict__ wdbf,  // H x H
    const float* __restrict__ b_ih,
    const float* __restrict__ b_hh,
    const float* __restrict__ b_dec,
    float* __restrict__ pre,                  // B x G
    float* __restrict__ dec)                  // B x H
{
  const int wave = (int)threadIdx.x >> 5;
  const int lane = (int)threadIdx.x & 31;
  const int half = lane >> 4;
  const int l16  = lane & 15;
  const int tile = (int)blockIdx.x * 8 + wave;

  v8f c00 = {}, c01 = {}, c10 = {}, c11 = {};

  if (tile < 512) {
    // ---- gates: 8 M-blocks x 64 N-blocks of 32x32 ----
    const int tM = tile & 7, tN = tile >> 3;
    const int row0 = tM * 32 + l16;
    const int j0   = tN * 32 + l16, j1 = j0 + 16;
    const unsigned short* ax0 = xibf + (size_t)row0 * F_DIM;
    const unsigned short* ax1 = ax0 + 16 * F_DIM;
    const unsigned short* ah0 = hbf + (size_t)row0 * H_DIM;
    const unsigned short* ah1 = ah0 + 16 * H_DIM;
    const unsigned short* w0  = wg + (size_t)j0 * KG;
    const unsigned short* w1  = wg + (size_t)j1 * KG;

    for (int k0 = 0; k0 < F_DIM; k0 += 32) {            // xi part of K
      const int kA1 = k0 + half * 16, kA2 = kA1 + 8, kB = k0 + half * 16;
      v16bf a0 = ld16(ax0, kA1, kA2), a1 = ld16(ax1, kA1, kA2);
      v16bf b0 = ld16(w0, kB, kB + 8), b1 = ld16(w1, kB, kB + 8);
      c00 = WMMA_BF16(a0, b0, c00);  c01 = WMMA_BF16(a0, b1, c01);
      c10 = WMMA_BF16(a1, b0, c10);  c11 = WMMA_BF16(a1, b1, c11);
    }
    const unsigned short* wh0 = w0 + F_DIM;             // w_hh part of rows
    const unsigned short* wh1 = w1 + F_DIM;
    for (int k0 = 0; k0 < H_DIM; k0 += 32) {            // h part of K
      const int kA1 = k0 + half * 16, kA2 = kA1 + 8, kB = k0 + half * 16;
      v16bf a0 = ld16(ah0, kA1, kA2), a1 = ld16(ah1, kA1, kA2);
      v16bf b0 = ld16(wh0, kB, kB + 8), b1 = ld16(wh1, kB, kB + 8);
      c00 = WMMA_BF16(a0, b0, c00);  c01 = WMMA_BF16(a0, b1, c01);
      c10 = WMMA_BF16(a1, b0, c10);  c11 = WMMA_BF16(a1, b1, c11);
    }
    const float bi0 = b_ih[j0] + b_hh[j0];
    const float bi1 = b_ih[j1] + b_hh[j1];
    #pragma unroll
    for (int r = 0; r < 8; ++r) {                       // C: VGPR r -> M = r + 8*half
      const int m0 = tM * 32 + half * 8 + r, m1 = m0 + 16;
      pre[(size_t)m0 * G_DIM + j0] = c00[r] + bi0;
      pre[(size_t)m0 * G_DIM + j1] = c01[r] + bi1;
      pre[(size_t)m1 * G_DIM + j0] = c10[r] + bi0;
      pre[(size_t)m1 * G_DIM + j1] = c11[r] + bi1;
    }
  } else {
    // ---- decomp: 8 M-blocks x 16 N-blocks of 32x32 ----
    const int t2 = tile - 512;
    const int tM = t2 & 7, tN = t2 >> 3;
    const int row0 = tM * 32 + l16;
    const int j0   = tN * 32 + l16, j1 = j0 + 16;
    const unsigned short* ac0 = cbf + (size_t)row0 * H_DIM;
    const unsigned short* ac1 = ac0 + 16 * H_DIM;
    const unsigned short* w0  = wdbf + (size_t)j0 * H_DIM;
    const unsigned short* w1  = wdbf + (size_t)j1 * H_DIM;

    for (int k0 = 0; k0 < H_DIM; k0 += 32) {
      const int kA1 = k0 + half * 16, kA2 = kA1 + 8, kB = k0 + half * 16;
      v16bf a0 = ld16(ac0, kA1, kA2), a1 = ld16(ac1, kA1, kA2);
      v16bf b0 = ld16(w0, kB, kB + 8), b1 = ld16(w1, kB, kB + 8);
      c00 = WMMA_BF16(a0, b0, c00);  c01 = WMMA_BF16(a0, b1, c01);
      c10 = WMMA_BF16(a1, b0, c10);  c11 = WMMA_BF16(a1, b1, c11);
    }
    const float bd0 = b_dec[j0], bd1 = b_dec[j1];
    #pragma unroll
    for (int r = 0; r < 8; ++r) {
      const int m0 = tM * 32 + half * 8 + r, m1 = m0 + 16;
      dec[(size_t)m0 * H_DIM + j0] = fast_tanh(c00[r] + bd0);
      dec[(size_t)m0 * H_DIM + j1] = fast_tanh(c01[r] + bd1);
      dec[(size_t)m1 * H_DIM + j0] = fast_tanh(c10[r] + bd0);
      dec[(size_t)m1 * H_DIM + j1] = fast_tanh(c11[r] + bd1);
    }
  }
}

// ---------------------------------------------------------------------------
// Per-step pointwise LSTM update (B*H elements). Also emits bf16 copies of
// the new h/c state so the next step's GEMM loads fragments with no cvt.
// ---------------------------------------------------------------------------
__global__ __launch_bounds__(256) void tplstm_pointwise(
    const float* __restrict__ ip_s,    // B x F1 (t at column 256)
    const float* __restrict__ cprev,   // B x H, f32
    const float* __restrict__ pre,     // B x G
    const float* __restrict__ dec,     // B x H (C_ST, tanh applied)
    float* __restrict__ hs_s, float* __restrict__ cs_s,
    unsigned short* __restrict__ hbf, unsigned short* __restrict__ cbf,
    float* __restrict__ hT, float* __restrict__ cT, int last)
{
  const int idx = (int)blockIdx.x * 256 + (int)threadIdx.x;  // < B*H
  const int b = idx >> 9, hh = idx & 511;

  const float t   = ip_s[(size_t)b * F1_DIM + F_DIM];
  const float T   = 1.0f / __logf(t + 2.7183f);
  const float cst = dec[idx];
  const float cpt = cprev[idx] + (T - 1.0f) * cst;

  const float* pr = pre + (size_t)b * G_DIM;
  const float ig = fast_sigmoid(pr[hh]);
  const float fg = fast_sigmoid(pr[H_DIM + hh]);
  const float gg = fast_tanh(pr[2 * H_DIM + hh]);
  const float og = fast_sigmoid(pr[3 * H_DIM + hh]);

  const float ct = fg * cpt + ig * gg;
  const float ht = og * fast_tanh(ct);

  hs_s[idx] = ht;
  cs_s[idx] = ct;
  hbf[idx] = f2bf(ht);
  cbf[idx] = f2bf(ct);
  if (last) { hT[idx] = ht; cT[idx] = ct; }
}

// ---------------------------------------------------------------------------
extern "C" void kernel_launch(void* const* d_in, const int* in_sizes, int n_in,
                              void* d_out, int out_size, void* d_ws, size_t ws_size,
                              hipStream_t stream) {
  const float* ip    = (const float*)d_in[0];
  const float* h0    = (const float*)d_in[1];
  const float* c0    = (const float*)d_in[2];
  const float* w_ih  = (const float*)d_in[3];
  const float* w_hh  = (const float*)d_in[4];
  const float* w_dec = (const float*)d_in[5];
  const float* b_ih  = (const float*)d_in[6];
  const float* b_hh  = (const float*)d_in[7];
  const float* b_dec = (const float*)d_in[8];

  // Output tuple: hs (S*B*H), hT (B*H), cT (B*H), cs (S*B*H)
  float* hs = (float*)d_out;
  float* hT = hs + (size_t)S_DIM * B_DIM * H_DIM;
  float* cT = hT + (size_t)B_DIM * H_DIM;
  float* cs = cT + (size_t)B_DIM * H_DIM;

  // Workspace layout (~40 MB)
  float* pre = (float*)d_ws;                                    // B*G f32
  float* dec = pre + (size_t)B_DIM * G_DIM;                     // B*H f32
  unsigned short* xibf = (unsigned short*)(dec + (size_t)B_DIM * H_DIM); // S*B*F
  unsigned short* wg   = xibf + (size_t)S_DIM * B_DIM * F_DIM;  // G*KG
  unsigned short* wdbf = wg   + (size_t)G_DIM * KG;             // H*H
  unsigned short* hbf  = wdbf + (size_t)H_DIM * H_DIM;          // B*H
  unsigned short* cbf  = hbf  + (size_t)B_DIM * H_DIM;          // B*H

  // ---- one-time prologue: pack everything static to bf16 (off the chain) ----
  {
    int n;
    n = S_DIM * B_DIM * F_DIM;   // xi for ALL steps (drop time column)
    conv_bf16<<<n / 256, 256, 0, stream>>>(ip, xibf, n, F1_DIM, F_DIM, F_DIM, 0);
    n = G_DIM * F_DIM;           // w_ih into wg[:, 0:256]
    conv_bf16<<<n / 256, 256, 0, stream>>>(w_ih, wg, n, F_DIM, F_DIM, KG, 0);
    n = G_DIM * H_DIM;           // w_hh into wg[:, 256:768]
    conv_bf16<<<n / 256, 256, 0, stream>>>(w_hh, wg, n, H_DIM, H_DIM, KG, F_DIM);
    n = H_DIM * H_DIM;           // w_decomp
    conv_bf16<<<n / 256, 256, 0, stream>>>(w_dec, wdbf, n, H_DIM, H_DIM, H_DIM, 0);
    n = B_DIM * H_DIM;           // initial state
    conv_bf16<<<n / 256, 256, 0, stream>>>(h0, hbf, n, H_DIM, H_DIM, H_DIM, 0);
    conv_bf16<<<n / 256, 256, 0, stream>>>(c0, cbf, n, H_DIM, H_DIM, H_DIM, 0);
  }

  const int gemm_blocks = (512 + 128) / 8;        // 80 blocks x 8 waves, exact
  const int pw_blocks   = (B_DIM * H_DIM) / 256;  // 512

  for (int s = 0; s < S_DIM; ++s) {
    const float* ip_s = ip + (size_t)s * B_DIM * F1_DIM;
    const float* cp   = s ? cs + (size_t)(s - 1) * B_DIM * H_DIM : c0;

    tplstm_gemm<<<gemm_blocks, 256, 0, stream>>>(
        xibf + (size_t)s * B_DIM * F_DIM, hbf, cbf, wg, wdbf,
        b_ih, b_hh, b_dec, pre, dec);

    tplstm_pointwise<<<pw_blocks, 256, 0, stream>>>(
        ip_s, cp, pre, dec,
        hs + (size_t)s * B_DIM * H_DIM,
        cs + (size_t)s * B_DIM * H_DIM,
        hbf, cbf, hT, cT, (s == S_DIM - 1) ? 1 : 0);
  }
}
